// MultiHeadSelfAttention_35673998361023
// MI455X (gfx1250) — matrix operand order repaired and checked by reading, not compiled, not run on hardware
//
#include <hip/hip_runtime.h>
#include <stdint.h>

#define T_SEQ 4096
#define D_MODEL 1024
#define NH 16
#define DHEAD 64

typedef __attribute__((ext_vector_type(16))) __bf16 v16bf;
typedef __attribute__((ext_vector_type(8)))  float  v8f;
typedef __attribute__((__vector_size__(4 * sizeof(int)))) int vi4;

// ---------- helpers ----------

static __device__ __forceinline__ unsigned short f2bf(float f) {
  unsigned int u = __float_as_uint(f);
  u += 0x7FFFu + ((u >> 16) & 1u);        // round-to-nearest-even
  return (unsigned short)(u >> 16);
}

// A fragment (16x32, 16-bit): lane half kh: K in [8*kh, 8*kh+8) and [16+8*kh, +8)
static __device__ __forceinline__ v16bf frag_a(const unsigned short* row, int kh) {
  union { uint4 u[2]; v16bf v; } f;
  f.u[0] = *(const uint4*)(row + 8 * kh);
  f.u[1] = *(const uint4*)(row + 8 * kh + 16);
  return f.v;
}

// B fragment (32x16, 16-bit, LDS stored [N][K]): lane half kh: K in [8*kh, 8*kh+8) and [16+8*kh, +8)
static __device__ __forceinline__ v16bf frag_b(const unsigned short* row, int kh) {
  union { uint4 u[2]; v16bf v; } f;
  f.u[0] = *(const uint4*)(row + 8 * kh);
  f.u[1] = *(const uint4*)(row + 8 * kh + 16);
  return f.v;
}

static __device__ __forceinline__ v8f wmma_bf16(v16bf a, v16bf b, v8f c) {
  return __builtin_amdgcn_wmma_f32_16x16x32_bf16(false, a, false, b, (short)0, c,
                                                 false, false);
}

// ---------- async global->LDS copy (16B per lane), with sync fallback ----------

#if __has_builtin(__builtin_amdgcn_global_load_async_to_lds_b128)
static __device__ __forceinline__ void async_cp16(const unsigned short* g, unsigned short* l) {
  __builtin_amdgcn_global_load_async_to_lds_b128(
      (__attribute__((address_space(1))) vi4*)g,
      (__attribute__((address_space(3))) vi4*)l, 0, 0);
}
static __device__ __forceinline__ void async_wait0() {
#if __has_builtin(__builtin_amdgcn_s_wait_asynccnt)
  __builtin_amdgcn_s_wait_asynccnt(0);
#else
  asm volatile("s_wait_asynccnt 0" ::: "memory");
#endif
}
#else
static __device__ __forceinline__ void async_cp16(const unsigned short* g, unsigned short* l) {
  *(uint4*)l = *(const uint4*)g;
}
static __device__ __forceinline__ void async_wait0() {}
#endif

// ---------- f32 -> bf16 conversion ----------

__global__ void cvt_f32_bf16(const float* __restrict__ s, unsigned short* __restrict__ d, int n) {
  int i = blockIdx.x * blockDim.x + threadIdx.x;
  if (i < n) d[i] = f2bf(s[i]);
}

// ---------- GEMM: C[M=4096, N] = A[4096,K]bf16 @ B[K,N]bf16 ----------
// MODE 0: scatter bf16 into Q/K [H][T][64], V [H][64][T]   MODE 1: f32 out [M][N]
template <int MODE>
__global__ __launch_bounds__(256) void gemm_bf16(const unsigned short* __restrict__ A,
                                                 const unsigned short* __restrict__ B,
                                                 int N, int K, float* __restrict__ outf,
                                                 unsigned short* __restrict__ qd,
                                                 unsigned short* __restrict__ kd,
                                                 unsigned short* __restrict__ vd) {
  __shared__ __align__(16) unsigned short As[128 * 72];   // [m][k] stride 72
  __shared__ __align__(16) unsigned short Bs[128 * 72];   // [n][k] stride 72 (transposed)

  const int tid  = threadIdx.x;
  const int w    = tid >> 5;
  const int lane = tid & 31;
  const int ml   = lane & 15;
  const int kh   = lane >> 4;
  const int mw   = w & 3;    // 4 row-groups of 32
  const int nw   = w >> 2;   // 2 col-groups of 64
  const int mbase = blockIdx.x * 128;
  const int nbase = blockIdx.y * 128;

  const v8f zacc = {0.f, 0.f, 0.f, 0.f, 0.f, 0.f, 0.f, 0.f};
  v8f acc[2][4];
#pragma unroll
  for (int mi = 0; mi < 2; ++mi)
#pragma unroll
    for (int ni = 0; ni < 4; ++ni) acc[mi][ni] = zacc;

  for (int k0 = 0; k0 < K; k0 += 64) {
    // A tile 128x64: async straight into LDS (row-major matches A-frag layout)
    const unsigned short* Ag = A + mbase * K + k0;
#pragma unroll
    for (int it = 0; it < 4; ++it) {
      int idx = it * 256 + tid;
      int r  = idx >> 3;
      int c8 = (idx & 7) * 8;
      async_cp16(Ag + r * K + c8, &As[r * 72 + c8]);
    }
    // B tile: 64(k) x 128(n), stored transposed [n][k]
    const unsigned short* Bg = B + k0 * N + nbase;
#pragma unroll
    for (int it = 0; it < 4; ++it) {
      int idx = it * 256 + tid;
      int kk = idx >> 4;
      int n8 = (idx & 15) * 8;
      uint4 t = *(const uint4*)(Bg + kk * N + n8);
      const unsigned short* ts = (const unsigned short*)&t;
#pragma unroll
      for (int i = 0; i < 8; ++i) Bs[(n8 + i) * 72 + kk] = ts[i];
    }
    if (k0 + 64 < K) {  // prefetch next B tile into L2
      const unsigned short* Bn = B + (k0 + 64) * N + nbase;
      __builtin_prefetch(Bn + (tid >> 4) * N + (tid & 15) * 8, 0, 1);
    }
    async_wait0();
    __syncthreads();

#pragma unroll
    for (int kc = 0; kc < 64; kc += 32) {
      v16bf af[2], bf_[4];
#pragma unroll
      for (int mi = 0; mi < 2; ++mi)
        af[mi] = frag_a(&As[(mw * 32 + mi * 16 + ml) * 72 + kc], kh);
#pragma unroll
      for (int ni = 0; ni < 4; ++ni)
        bf_[ni] = frag_b(&Bs[(nw * 64 + ni * 16 + ml) * 72 + kc], kh);
#pragma unroll
      for (int mi = 0; mi < 2; ++mi)
#pragma unroll
        for (int ni = 0; ni < 4; ++ni)
          acc[mi][ni] = wmma_bf16(af[mi], bf_[ni], acc[mi][ni]);
    }
    __syncthreads();
  }

  // epilogue
#pragma unroll
  for (int mi = 0; mi < 2; ++mi) {
#pragma unroll
    for (int ni = 0; ni < 4; ++ni) {
      int trow0 = mbase + mw * 32 + mi * 16 + 8 * kh;
      int col   = nbase + nw * 64 + ni * 16 + ml;
      if (MODE == 0) {
        int seg = col >> 10;
        int c   = col & 1023;
        int hh  = c >> 6;
        int dh  = c & 63;
        if (seg == 2) {  // V stored [H][DH][T] so attention can async-copy rows
#pragma unroll
          for (int r = 0; r < 8; ++r)
            vd[(hh * DHEAD + dh) * T_SEQ + trow0 + r] = f2bf(acc[mi][ni][r]);
        } else {
          unsigned short* dst = (seg == 0) ? qd : kd;
#pragma unroll
          for (int r = 0; r < 8; ++r)
            dst[(hh * T_SEQ + trow0 + r) * DHEAD + dh] = f2bf(acc[mi][ni][r]);
        }
      } else {
#pragma unroll
        for (int r = 0; r < 8; ++r)
          outf[(trow0 + r) * D_MODEL + col] = acc[mi][ni][r];
      }
    }
  }
}

// ---------- flash attention (causal), one head per blockIdx.y ----------
__global__ __launch_bounds__(256) void attn_kernel(const unsigned short* __restrict__ Qb,
                                                   const unsigned short* __restrict__ Kb,
                                                   const unsigned short* __restrict__ Vb,
                                                   unsigned short* __restrict__ Ob) {
  __shared__ __align__(16) unsigned short Qs[128 * 72];        // [qrow][dh]
  __shared__ __align__(16) unsigned short Ks[2][32 * 72];      // [key][dh]  (B layout for S)
  __shared__ __align__(16) unsigned short Vt[2][64 * 40];      // [dh][key]  (B layout for P@V)
  __shared__ __align__(16) unsigned short Ps[8 * 16 * 40];     // per-wave P [16][32] stride 40

  const int tid  = threadIdx.x;
  const int w    = tid >> 5;
  const int lane = tid & 31;
  const int ml   = lane & 15;
  const int kh   = lane >> 4;
  const int h     = blockIdx.y;
  const int qbase = blockIdx.x * 128;

  const unsigned short* Qh = Qb + h * T_SEQ * DHEAD;   // [T][64]
  const unsigned short* Kh = Kb + h * T_SEQ * DHEAD;   // [T][64]
  const unsigned short* Vh = Vb + h * DHEAD * T_SEQ;   // [64][T]

  const int rowb = qbase + w * 16 + 8 * kh;  // this lane's first query row
  const int nkt  = qbase / 32 + 4;           // causal tile bound for block

  // per-thread addresses for K/V tile fills (16B per lane)
  const int kv_kr = tid >> 3, kv_c8 = (tid & 7) * 8;   // K: [key][dh]
  const int v_dh  = tid >> 2, v_q8  = (tid & 3) * 8;   // V: [dh][key]

  // Q tile (128x64) async into LDS
#pragma unroll
  for (int it = 0; it < 4; ++it) {
    int idx = it * 256 + tid;
    int r  = idx >> 3;
    int c8 = (idx & 7) * 8;
    async_cp16(Qh + (qbase + r) * DHEAD + c8, &Qs[r * 72 + c8]);
  }
  // preload K/V tile 0 into buffer 0
  async_cp16(Kh + kv_kr * DHEAD + kv_c8, &Ks[0][kv_kr * 72 + kv_c8]);
  async_cp16(Vh + v_dh * T_SEQ + v_q8,   &Vt[0][v_dh * 40 + v_q8]);
  async_wait0();
  __syncthreads();

  const v8f zacc = {0.f, 0.f, 0.f, 0.f, 0.f, 0.f, 0.f, 0.f};
  v8f o[4];
#pragma unroll
  for (int ni = 0; ni < 4; ++ni) o[ni] = zacc;
  float mrow[8], lrow[8];
#pragma unroll
  for (int r = 0; r < 8; ++r) { mrow[r] = -1e30f; lrow[r] = 0.f; }

  for (int kt = 0; kt < nkt; ++kt) {
    const int kb  = kt * 32;
    const int cur = kt & 1;
    // double-buffer: issue async copies for the next tile while computing this one
    if (kt + 1 < nkt) {
      const int kn = (kt + 1) * 32;
      async_cp16(Kh + (kn + kv_kr) * DHEAD + kv_c8, &Ks[cur ^ 1][kv_kr * 72 + kv_c8]);
      async_cp16(Vh + v_dh * T_SEQ + kn + v_q8,     &Vt[cur ^ 1][v_dh * 40 + v_q8]);
    }

    if (kb <= qbase + w * 16 + 15) {  // wave has at least one unmasked row
      // S = Q(16x64) @ K^T(64x32) -> two 16x16 tiles
      v8f s0 = zacc, s1 = zacc;
#pragma unroll
      for (int kc = 0; kc < 64; kc += 32) {
        v16bf aq = frag_a(&Qs[(w * 16 + ml) * 72 + kc], kh);
        v16bf b0 = frag_b(&Ks[cur][(ml)      * 72 + kc], kh);
        v16bf b1 = frag_b(&Ks[cur][(16 + ml) * 72 + kc], kh);
        s0 = wmma_bf16(aq, b0, s0);
        s1 = wmma_bf16(aq, b1, s1);
      }

      const int key0 = kb + ml;
      const int key1 = kb + 16 + ml;
      float rm[8];
#pragma unroll
      for (int r = 0; r < 8; ++r) {
        float v0 = s0[r] * 0.125f, v1 = s1[r] * 0.125f;  // 1/sqrt(64)
        s0[r] = v0; s1[r] = v1;
        int qr = rowb + r;
        float t0 = (key0 <= qr) ? v0 : -1e30f;
        float t1 = (key1 <= qr) ? v1 : -1e30f;
        rm[r] = fmaxf(t0, t1);
      }
#pragma unroll
      for (int r = 0; r < 8; ++r) {
        float v = rm[r];
#pragma unroll
        for (int off = 1; off < 16; off <<= 1) v = fmaxf(v, __shfl_xor(v, off, 32));
        rm[r] = v;
      }
      float al[8], ps[8];
#pragma unroll
      for (int r = 0; r < 8; ++r) {
        float mn = fmaxf(mrow[r], rm[r]);
        al[r] = __expf(mrow[r] - mn);
        mrow[r] = mn;
      }
      unsigned short* Pw = &Ps[w * 16 * 40];
#pragma unroll
      for (int r = 0; r < 8; ++r) {
        int qr = rowb + r;
        float p0 = (key0 <= qr) ? __expf(s0[r] - mrow[r]) : 0.f;
        float p1 = (key1 <= qr) ? __expf(s1[r] - mrow[r]) : 0.f;
        ps[r] = p0 + p1;
        Pw[(r + 8 * kh) * 40 + ml]      = f2bf(p0);
        Pw[(r + 8 * kh) * 40 + 16 + ml] = f2bf(p1);
      }
#pragma unroll
      for (int r = 0; r < 8; ++r) {
        float v = ps[r];
#pragma unroll
        for (int off = 1; off < 16; off <<= 1) v += __shfl_xor(v, off, 32);
        lrow[r] = lrow[r] * al[r] + v;
      }
#pragma unroll
      for (int ni = 0; ni < 4; ++ni)
#pragma unroll
        for (int r = 0; r < 8; ++r) o[ni][r] *= al[r];

      // wave-private LDS P: wait for our own ds stores, then re-load as A fragment
      asm volatile("s_wait_dscnt 0" ::: "memory");
      v16bf ap = frag_a(&Pw[ml * 40], kh);
#pragma unroll
      for (int ni = 0; ni < 4; ++ni) {
        v16bf bv = frag_b(&Vt[cur][(ni * 16 + ml) * 40], kh);
        o[ni] = wmma_bf16(ap, bv, o[ni]);
      }
    }
    async_wait0();
    __syncthreads();
  }

  // normalize + store O as bf16 [T][D] with per-head column offset
#pragma unroll
  for (int ni = 0; ni < 4; ++ni) {
#pragma unroll
    for (int r = 0; r < 8; ++r) {
      int qr  = rowb + r;
      int col = h * DHEAD + ni * 16 + ml;
      Ob[qr * D_MODEL + col] = f2bf(o[ni][r] / lrow[r]);
    }
  }
}

// ---------- launch ----------

extern "C" void kernel_launch(void* const* d_in, const int* in_sizes, int n_in,
                              void* d_out, int out_size, void* d_ws, size_t ws_size,
                              hipStream_t stream) {
  (void)in_sizes; (void)n_in; (void)out_size; (void)ws_size;
  const float* x    = (const float*)d_in[0];
  const float* wqkv = (const float*)d_in[1];
  const float* wo   = (const float*)d_in[2];
  float* out = (float*)d_out;

  char* ws = (char*)d_ws;
  const size_t MB = (size_t)1 << 20;
  unsigned short* xbf = (unsigned short*)(ws + 0 * MB);   // 8 MiB
  unsigned short* wqb = (unsigned short*)(ws + 8 * MB);   // 6 MiB
  unsigned short* wob = (unsigned short*)(ws + 14 * MB);  // 2 MiB
  unsigned short* Qb  = (unsigned short*)(ws + 16 * MB);  // 8 MiB [H][T][64]
  unsigned short* Kb  = (unsigned short*)(ws + 24 * MB);  // 8 MiB [H][T][64]
  unsigned short* Vb  = (unsigned short*)(ws + 32 * MB);  // 8 MiB [H][64][T]
  unsigned short* Ob  = (unsigned short*)(ws + 40 * MB);  // 8 MiB [T][D]

  const int nx = T_SEQ * D_MODEL;
  const int nq = D_MODEL * 3 * D_MODEL;
  const int nw = D_MODEL * D_MODEL;
  cvt_f32_bf16<<<(nx + 255) / 256, 256, 0, stream>>>(x, xbf, nx);
  cvt_f32_bf16<<<(nq + 255) / 256, 256, 0, stream>>>(wqkv, wqb, nq);
  cvt_f32_bf16<<<(nw + 255) / 256, 256, 0, stream>>>(wo, wob, nw);

  // QKV projection: [4096,1024] @ [1024,3072] -> per-head bf16 Q/K/V
  gemm_bf16<0><<<dim3(32, 24), 256, 0, stream>>>(xbf, wqb, 3 * D_MODEL, D_MODEL,
                                                 nullptr, Qb, Kb, Vb);
  // causal flash attention
  attn_kernel<<<dim3(T_SEQ / 128, NH), 256, 0, stream>>>(Qb, Kb, Vb, Ob);
  // output projection: [4096,1024] @ [1024,1024] -> f32 out
  gemm_bf16<1><<<dim3(32, 8), 256, 0, stream>>>(Ob, wob, D_MODEL, D_MODEL,
                                                out, nullptr, nullptr, nullptr);
}
